// NuggetScorer_9311489098362
// MI455X (gfx1250) — matrix operand order repaired and checked by reading, not compiled, not run on hardware
//
#include <hip/hip_runtime.h>
#include <stdint.h>

#define Bn 8
#define Sn 4096
#define Dn 2048
#define Mn 410
#define APITCH (Dn + 8)          // bf16 elements per LDS row (+16B pad, kills bank conflicts)

#define K1_ROWS 64               // scorer rows per block (4 row-tiles)
#define K1_RT   4
#define K4_ROWS 32               // value rows per block (2 row-tiles)
#define K4_RT   2
#define K4_BLK_PER_B ((Mn + K4_ROWS - 1) / K4_ROWS)   // 13

typedef __attribute__((ext_vector_type(16))) __bf16 v16bf;
typedef __attribute__((ext_vector_type(8)))  float  v8f;

union Frag16 { v16bf v; uint4 q[2]; };

__device__ __forceinline__ unsigned short f2bf(float f) {
  unsigned u = __builtin_bit_cast(unsigned, f);
  u += 0x7fffu + ((u >> 16) & 1u);           // round-to-nearest-even
  return (unsigned short)(u >> 16);
}

__device__ __forceinline__ v8f wmma_bf16(const Frag16& a, const Frag16& b, v8f c) {
  return __builtin_amdgcn_wmma_f32_16x16x32_bf16(false, a.v, false, b.v,
                                                 (short)0, c, false, false);
}

// ---------------- K0: transpose + convert weights to bf16 [e][d] ----------------
__global__ void __launch_bounds__(256)
transpose_bf16(const float* __restrict__ src, unsigned short* __restrict__ dst) {
  __shared__ unsigned short tile[32][33];
  const int tx = threadIdx.x & 31;
  const int tg = threadIdx.x >> 5;           // 0..7
  const int bx = blockIdx.x & 63;            // e-tile
  const int by = blockIdx.x >> 6;            // d-tile
  const int c0 = bx * 32, r0 = by * 32;
  for (int i = 0; i < 4; ++i) {
    int r = r0 + tg * 4 + i;                 // d
    int c = c0 + tx;                         // e (coalesced read)
    tile[tx][tg * 4 + i] = f2bf(src[(size_t)r * Dn + c]);
  }
  __syncthreads();
  for (int i = 0; i < 4; ++i) {
    int e = c0 + tg * 4 + i;
    int d = r0 + tx;                         // coalesced write
    dst[(size_t)e * Dn + d] = tile[tg * 4 + i][tx];
  }
}

// ---------------- K1: fused scorer GEMM: score = w2 . relu(X*W1 + b1) ----------------
// 64 rows/block; each wave owns 16 e-tiles and reuses each B fragment across
// 4 A row-tiles (4x L2-traffic reduction on W1t). Fragment loads are grouped
// ahead of the 4 back-to-back WMMAs so the scheduler can double-buffer.
__global__ void __launch_bounds__(256)
scorer_gemm(const float* __restrict__ X, const unsigned short* __restrict__ W1t,
            const float* __restrict__ b1, const float* __restrict__ w2,
            const float* __restrict__ b2, float* __restrict__ scores_out) {
  extern __shared__ char smem[];
  unsigned short* Ab = (unsigned short*)smem;
  float* ssc = (float*)(smem + (size_t)K1_ROWS * APITCH * 2);

  const int tid  = threadIdx.x;
  const int wave = tid >> 5;
  const int lane = tid & 31;
  const int half = lane >> 4;
  const int l16  = lane & 15;
  const size_t row0 = (size_t)blockIdx.x * K1_ROWS;

  if (tid < K1_ROWS) ssc[tid] = 0.0f;

  // stage 64 x 2048 f32 rows as bf16 into LDS
  const float* src = X + row0 * Dn;
  for (int i = tid; i < K1_ROWS * Dn / 4; i += 256) {
    int r = (i * 4) / Dn, c = (i * 4) % Dn;
    float4 v = *(const float4*)(src + (size_t)r * Dn + c);
    uint2 p;
    p.x = (unsigned)f2bf(v.x) | ((unsigned)f2bf(v.y) << 16);
    p.y = (unsigned)f2bf(v.z) | ((unsigned)f2bf(v.w) << 16);
    *(uint2*)&Ab[r * APITCH + c] = p;
  }
  __syncthreads();

  // per-row-tile LDS base pointers (so +d0 fits the 16-bit DS offset field)
  const unsigned short* ap0 = Ab + (0 * 16 + l16) * APITCH + half * 8;
  const unsigned short* ap1 = Ab + (1 * 16 + l16) * APITCH + half * 8;
  const unsigned short* ap2 = Ab + (2 * 16 + l16) * APITCH + half * 8;
  const unsigned short* ap3 = Ab + (3 * 16 + l16) * APITCH + half * 8;

  float part[K1_RT][8];
  for (int t = 0; t < K1_RT; ++t)
    for (int r = 0; r < 8; ++r) part[t][r] = 0.0f;

  for (int et = wave; et < Dn / 16; et += 8) {
    const int e0 = et * 16;
    v8f acc0 = {}, acc1 = {}, acc2 = {}, acc3 = {};
    const unsigned short* bp = W1t + (size_t)(e0 + l16) * Dn + half * 8;
#pragma unroll 2
    for (int d0 = 0; d0 < Dn; d0 += 32) {
      Frag16 b, a0, a1, a2, a3;
      // B 32x16 bf16: lane col n=l16, k = half*8+0..7, then 16+half*8+0..7 in W1t[e][d]
      b.q[0] = *(const uint4*)(bp + d0);
      b.q[1] = *(const uint4*)(bp + d0 + 16);
      // A 16x32 bf16: v0..3 -> k=half*8+0..7, v4..7 -> k=16+half*8+0..7
      a0.q[0] = *(const uint4*)(ap0 + d0);
      a0.q[1] = *(const uint4*)(ap0 + d0 + 16);
      a1.q[0] = *(const uint4*)(ap1 + d0);
      a1.q[1] = *(const uint4*)(ap1 + d0 + 16);
      a2.q[0] = *(const uint4*)(ap2 + d0);
      a2.q[1] = *(const uint4*)(ap2 + d0 + 16);
      a3.q[0] = *(const uint4*)(ap3 + d0);
      a3.q[1] = *(const uint4*)(ap3 + d0 + 16);
      acc0 = wmma_bf16(a0, b, acc0);
      acc1 = wmma_bf16(a1, b, acc1);
      acc2 = wmma_bf16(a2, b, acc2);
      acc3 = wmma_bf16(a3, b, acc3);
    }
    // epilogue: h = acc + b1[e]; score += w2[e]*relu(h)
    const float b1n = b1[e0 + l16];
    const float w2n = w2[e0 + l16];
    for (int r = 0; r < 8; ++r) {
      part[0][r] += w2n * fmaxf(acc0[r] + b1n, 0.0f);
      part[1][r] += w2n * fmaxf(acc1[r] + b1n, 0.0f);
      part[2][r] += w2n * fmaxf(acc2[r] + b1n, 0.0f);
      part[3][r] += w2n * fmaxf(acc3[r] + b1n, 0.0f);
    }
  }
  // C layout: row = r + 8*half (within row-tile t); reduce over the 16 n-lanes
  for (int t = 0; t < K1_RT; ++t)
    for (int r = 0; r < 8; ++r)
      atomicAdd(&ssc[t * 16 + half * 8 + r], part[t][r]);
  __syncthreads();
  if (tid < K1_ROWS)
    scores_out[row0 + tid] = ssc[tid] + b2[0];
}

// ---------------- K2: masked fill + n_nugget ----------------
__global__ void __launch_bounds__(256)
mask_scores(const int* __restrict__ attn, float* __restrict__ scores,
            int* __restrict__ nnug) {
  __shared__ int cnt;
  const int b = blockIdx.x;
  const int tid = threadIdx.x;
  if (tid == 0) cnt = 0;
  __syncthreads();
  int local = 0;
  for (int s = tid; s < Sn; s += 256) {
    int mk = attn[(size_t)b * Sn + s];
    local += mk;
    float v = scores[(size_t)b * Sn + s];
    scores[(size_t)b * Sn + s] = mk ? v : -3.402823466e38f;
  }
  atomicAdd(&cnt, local);
  __syncthreads();
  if (tid == 0) {
    int nt = cnt;
    int nn = (int)ceilf((float)nt * 0.1f);
    nn = nn < 1 ? 1 : nn;
    nn = nn > nt ? nt : nn;
    nnug[b] = nn;
  }
}

// ---------------- K3: stable top-M selection matching double-argsort ----------------
__global__ void __launch_bounds__(512)
select_topk(const float* __restrict__ scores, const int* __restrict__ nnug,
            float* __restrict__ out_mask, float* __restrict__ out_nsc,
            float* __restrict__ out_idxf, int* __restrict__ idx_ws) {
  __shared__ float sv[Sn];
  __shared__ int   rk[Sn];
  const int b = blockIdx.x;
  const int tid = threadIdx.x;
  const int bd = blockDim.x;
  for (int i = tid; i < Sn; i += bd) sv[i] = scores[(size_t)b * Sn + i];
  __syncthreads();
  // rank in stable descending sort: #greater + #equal-with-smaller-index
  for (int i = tid; i < Sn; i += bd) {
    float si = sv[i];
    int r = 0;
    for (int j = 0; j < Sn; ++j) {
      float sj = sv[j];
      r += (sj > si) || (sj == si && j < i);
    }
    rk[i] = r;
  }
  __syncthreads();
  const int nn = nnug[b];
  for (int i = tid; i < Sn; i += bd) {
    int r = rk[i];
    int pos = -1;
    if (r < nn) {               // valid nugget: placed in ascending token order
      pos = 0;
      for (int j = 0; j < i; ++j) pos += (rk[j] < nn);
    } else if (r < Mn) {        // overflow slot keeps its descending-score rank
      pos = r;
    }
    if (pos >= 0) {
      idx_ws[b * Mn + pos] = i;
      out_idxf[(size_t)b * Mn + pos] = (float)i;
      out_nsc[(size_t)b * Mn + pos] = sv[i];
    }
  }
  for (int k = tid; k < Mn; k += bd)
    out_mask[(size_t)b * Mn + k] = (k < nn) ? 1.0f : 0.0f;
}

// ---------------- K4: gather + value FFN GEMM ----------------
// 32 gathered rows/block (2 row-tiles); B fragment reused across both tiles.
__global__ void __launch_bounds__(256)
value_gemm(const float* __restrict__ H, const unsigned short* __restrict__ Wvt,
           const float* __restrict__ bv, const int* __restrict__ idx_ws,
           float* __restrict__ enc_out) {
  extern __shared__ char smem[];
  unsigned short* Ab = (unsigned short*)smem;
  int* sidx = (int*)(smem + (size_t)K4_ROWS * APITCH * 2);

  const int tid  = threadIdx.x;
  const int wave = tid >> 5;
  const int lane = tid & 31;
  const int half = lane >> 4;
  const int l16  = lane & 15;
  const int b    = blockIdx.x / K4_BLK_PER_B;
  const int k0   = (blockIdx.x % K4_BLK_PER_B) * K4_ROWS;

  if (tid < K4_ROWS) {
    int k = k0 + tid;
    sidx[tid] = (k < Mn) ? idx_ws[b * Mn + k] : 0;
  }
  __syncthreads();

  for (int i = tid; i < K4_ROWS * Dn / 4; i += 256) {
    int r = (i * 4) / Dn, c = (i * 4) % Dn;
    const float* src = H + ((size_t)b * Sn + sidx[r]) * Dn + c;
    float4 v = *(const float4*)src;
    uint2 p;
    p.x = (unsigned)f2bf(v.x) | ((unsigned)f2bf(v.y) << 16);
    p.y = (unsigned)f2bf(v.z) | ((unsigned)f2bf(v.w) << 16);
    *(uint2*)&Ab[r * APITCH + c] = p;
  }
  __syncthreads();

  const unsigned short* ap0 = Ab + (0 * 16 + l16) * APITCH + half * 8;
  const unsigned short* ap1 = Ab + (1 * 16 + l16) * APITCH + half * 8;

  for (int et = wave; et < Dn / 16; et += 8) {
    const int e0 = et * 16;
    v8f acc0 = {}, acc1 = {};
    const unsigned short* bp = Wvt + (size_t)(e0 + l16) * Dn + half * 8;
#pragma unroll 2
    for (int d0 = 0; d0 < Dn; d0 += 32) {
      Frag16 b, a0, a1;
      b.q[0] = *(const uint4*)(bp + d0);
      b.q[1] = *(const uint4*)(bp + d0 + 16);
      a0.q[0] = *(const uint4*)(ap0 + d0);
      a0.q[1] = *(const uint4*)(ap0 + d0 + 16);
      a1.q[0] = *(const uint4*)(ap1 + d0);
      a1.q[1] = *(const uint4*)(ap1 + d0 + 16);
      acc0 = wmma_bf16(a0, b, acc0);
      acc1 = wmma_bf16(a1, b, acc1);
    }
    const float bvn = bv[e0 + l16];
    for (int r = 0; r < 8; ++r) {
      int krow0 = k0 + 0 * 16 + half * 8 + r;
      int krow1 = k0 + 1 * 16 + half * 8 + r;
      if (krow0 < Mn)
        enc_out[((size_t)b * Mn + krow0) * Dn + (e0 + l16)] = acc0[r] + bvn;
      if (krow1 < Mn)
        enc_out[((size_t)b * Mn + krow1) * Dn + (e0 + l16)] = acc1[r] + bvn;
    }
  }
}

// ---------------- host launcher ----------------
extern "C" void kernel_launch(void* const* d_in, const int* in_sizes, int n_in,
                              void* d_out, int out_size, void* d_ws, size_t ws_size,
                              hipStream_t stream) {
  (void)in_sizes; (void)n_in; (void)out_size; (void)ws_size;
  const float* t_out = (const float*)d_in[0];
  const float* hid   = (const float*)d_in[1];
  const int*   attn  = (const int*)d_in[2];
  const float* W1    = (const float*)d_in[3];
  const float* b1    = (const float*)d_in[4];
  const float* w2    = (const float*)d_in[5];
  const float* b2    = (const float*)d_in[6];
  const float* Wv    = (const float*)d_in[7];
  const float* bv    = (const float*)d_in[8];

  float* out        = (float*)d_out;
  float* out_enc    = out;                                   // [B,M,D]
  float* out_mask   = out_enc + (size_t)Bn * Mn * Dn;        // [B,M]
  float* out_nsc    = out_mask + (size_t)Bn * Mn;            // [B,M]
  float* out_idxf   = out_nsc + (size_t)Bn * Mn;             // [B,M]
  float* out_scores = out_idxf + (size_t)Bn * Mn;            // [B,S]

  unsigned short* w1t = (unsigned short*)d_ws;               // 8.4 MB
  unsigned short* wvt = w1t + (size_t)Dn * Dn;               // 8.4 MB
  int* idx_ws = (int*)(wvt + (size_t)Dn * Dn);
  int* nnug   = idx_ws + Bn * Mn;

  transpose_bf16<<<64 * 64, 256, 0, stream>>>(W1, w1t);
  transpose_bf16<<<64 * 64, 256, 0, stream>>>(Wv, wvt);

  size_t lds1 = (size_t)K1_ROWS * APITCH * 2 + K1_ROWS * sizeof(float);
  scorer_gemm<<<Bn * Sn / K1_ROWS, 256, lds1, stream>>>(t_out, w1t, b1, w2, b2,
                                                        out_scores);

  mask_scores<<<Bn, 256, 0, stream>>>(attn, out_scores, nnug);

  select_topk<<<Bn, 512, 0, stream>>>(out_scores, nnug, out_mask, out_nsc,
                                      out_idxf, idx_ws);

  size_t lds4 = (size_t)K4_ROWS * APITCH * 2 + K4_ROWS * sizeof(int);
  value_gemm<<<Bn * K4_BLK_PER_B, 256, lds4, stream>>>(hid, wvt, bv, idx_ws,
                                                       out_enc);
}
